// TransformerEncoderLayer_48052094107995
// MI455X (gfx1250) — matrix operand order repaired and checked by reading, not compiled, not run on hardware
//
#include <hip/hip_runtime.h>
#include <cstdint>

#define DEV __device__ __forceinline__

typedef __attribute__((ext_vector_type(16))) __bf16 v16bf;
typedef __attribute__((ext_vector_type(8)))  float  v8f;
typedef unsigned short u16;

// problem dims (fixed by reference)
constexpr int Bb = 4, Ss = 2048, Dd = 1024, Hh = 16, DKk = 64, Ff = 4096;
constexpr int Mrows = Bb * Ss; // 8192

union Frag {
  v16bf v;
  uint4 q[2];
};

DEV uint4 ldg16(const u16* p) { return *reinterpret_cast<const uint4*>(p); }

DEV u16 f2bf(float f) {
  union { float f; unsigned u; } x; x.f = f;
  unsigned r = x.u + 0x7FFFu + ((x.u >> 16) & 1u);
  return (u16)(r >> 16);
}

DEV v8f wmma_bf16(const Frag& a, const Frag& b, v8f c) {
  return __builtin_amdgcn_wmma_f32_16x16x32_bf16(
      /*neg_a=*/false, a.v, /*neg_b=*/false, b.v,
      /*c_mod=*/(short)0, c, /*reuse_a=*/false, /*reuse_b=*/false);
}

// ---------------------------------------------------------------------------
// fp32 -> bf16 elementwise cast (activations)
// ---------------------------------------------------------------------------
__global__ void __launch_bounds__(256) cast_f32_bf16(const float* __restrict__ in,
                                                     u16* __restrict__ out, int n) {
  int i = blockIdx.x * 256 + threadIdx.x;
  if (i < n) out[i] = f2bf(in[i]);
}

// ---------------------------------------------------------------------------
// Batched tiled transpose + cast: in fp32 [batch][R][C] -> out bf16 [batch][C][R]
// block (32,8): one 32x32 tile per block through LDS.
// Weights become B^T [N,K] so WMMA B-fragments are contiguous per lane and the
// GEMM needs no LDS staging or barriers.
// ---------------------------------------------------------------------------
__global__ void __launch_bounds__(256) transpose_cast(
    const float* __restrict__ in, u16* __restrict__ out, int R, int C)
{
  __shared__ float tile[32][33];
  const float* ib = in + (long)blockIdx.z * R * C;
  u16* ob = out + (long)blockIdx.z * R * C;
  const int c0 = blockIdx.x * 32, r0 = blockIdx.y * 32;
  #pragma unroll
  for (int i = 0; i < 4; ++i)
    tile[threadIdx.y + i * 8][threadIdx.x] =
        ib[(long)(r0 + threadIdx.y + i * 8) * C + c0 + threadIdx.x];
  __syncthreads();
  #pragma unroll
  for (int i = 0; i < 4; ++i)
    ob[(long)(c0 + threadIdx.y + i * 8) * R + r0 + threadIdx.x] =
        f2bf(tile[threadIdx.x][threadIdx.y + i * 8]);
}

// ---------------------------------------------------------------------------
// bf16 WMMA GEMM, barrier-free: C[M,N] = A[M,K] * Bt[N,K]^T + bias
// Each wave owns a 32(M) x 64(N) tile: 2 M-subtiles x 4 N-subtiles = 8
// accumulators. Per 32-K step: 2 A fragments + 4 B fragments feed 8
// back-to-back WMMAs (B reused across M-subtiles, A across N-subtiles).
// Block = 8 waves stacked along M (covers 256 rows); no LDS, no barriers.
// MODE: 1 = bf16 out scattered to [B,H,S,DK]    (q, k projections)
//       4 = bf16 out scattered to [B,H,DK,S]    (v projection, transposed)
//       2 = bf16 out row-major with ReLU        (FFN hidden)
//       3 = fp32 out row-major                  (attn proj, FFN out)
// ---------------------------------------------------------------------------
template<int MODE>
__global__ void __launch_bounds__(256) gemm_bf16_wmma(
    const u16* __restrict__ A, int lda,
    const u16* __restrict__ Bt,           // [N, K] row-major (ldb = K)
    const float* __restrict__ bias,
    float* __restrict__ OutF, u16* __restrict__ OutB,
    int N, int K)
{
  const int tid  = threadIdx.x;
  const int wave = tid >> 5, lane = tid & 31;
  const int half = lane >> 4, lh = lane & 15;
  const int n0 = blockIdx.x * 64;
  const int m0 = (blockIdx.y * 8 + wave) * 32;
  const u16* Ap0 = A + (long)(m0 + lh) * lda;        // M-subtile 0 row
  const u16* Ap1 = A + (long)(m0 + 16 + lh) * lda;   // M-subtile 1 row
  const u16* bpt[4];
  #pragma unroll
  for (int t = 0; t < 4; ++t)
    bpt[t] = Bt + (long)(n0 + t * 16 + lh) * K;   // per-lane B row (n fixed, k contiguous)

  v8f acc[2][4] = {};
  for (int k0 = 0; k0 < K; k0 += 32) {
    Frag fa0, fa1;
    fa0.q[0] = ldg16(Ap0 + k0 + half * 8);        // elems j<8 : K = half*8 + j
    fa0.q[1] = ldg16(Ap0 + k0 + 16 + half * 8);   // elems j>=8: K = 16 + half*8 + j-8
    fa1.q[0] = ldg16(Ap1 + k0 + half * 8);
    fa1.q[1] = ldg16(Ap1 + k0 + 16 + half * 8);
    if (k0 + 32 < K) {
      __builtin_prefetch(Ap0 + k0 + 32, 0, 1);    // global_prefetch_b8
      __builtin_prefetch(Ap1 + k0 + 32, 0, 1);
    }
    #pragma unroll
    for (int t = 0; t < 4; ++t) {
      Frag fb;
      fb.q[0] = ldg16(bpt[t] + k0 + half * 8);    // elems j<8 : K = half*8 + j; j>=8: K = 16 + half*8 + j-8
      fb.q[1] = ldg16(bpt[t] + k0 + 16 + half * 8);
      acc[0][t] = wmma_bf16(fa0, fb, acc[0][t]);
      acc[1][t] = wmma_bf16(fa1, fb, acc[1][t]);
    }
  }

  #pragma unroll
  for (int mt = 0; mt < 2; ++mt) {
    #pragma unroll
    for (int t = 0; t < 4; ++t) {
      const int ng = n0 + t * 16 + lh;   // C layout: col = lane%16
      const float bs = bias[ng];
      #pragma unroll
      for (int r = 0; r < 8; ++r) {      // rows = r + 8*(lane/16)
        const int mg = m0 + mt * 16 + r + 8 * half;
        float o = acc[mt][t][r] + bs;
        if constexpr (MODE == 3) {
          OutF[(long)mg * N + ng] = o;
        } else if constexpr (MODE == 2) {
          OutB[(long)mg * N + ng] = f2bf(fmaxf(o, 0.f));
        } else {
          const int bi = mg >> 11, si = mg & (Ss - 1);      // S = 2048
          const int hi = ng >> 6,  kk = ng & (DKk - 1);     // DK = 64
          if constexpr (MODE == 1)
            OutB[(((long)(bi * Hh + hi)) * Ss + si) * DKk + kk] = f2bf(o);   // [B,H,S,DK]
          else  // MODE == 4
            OutB[(((long)(bi * Hh + hi)) * DKk + kk) * Ss + si] = f2bf(o);   // [B,H,DK,S]
        }
      }
    }
  }
}

// ---------------------------------------------------------------------------
// Flash attention: one wave = one 16-row query tile of one (b,h).
// Q:[B,H,S,DK] bf16   K:[B,H,S,DK] bf16   Vt:[B,H,DK,S] bf16
// 32-key blocks: 4 WMMAs for scores (2 n-tiles x 2 k-steps over DK=64),
// online softmax in fp32, P through LDS (C-layout -> A-layout), 4 WMMAs for PV.
// ---------------------------------------------------------------------------
__global__ void __launch_bounds__(256) attn_flash(
    const u16* __restrict__ Q, const u16* __restrict__ Km,
    const u16* __restrict__ Vt, u16* __restrict__ Ctx)
{
  __shared__ __align__(16) u16 ldsP[8][16 * 32];
  const int tid  = threadIdx.x;
  const int wave = tid >> 5, lane = tid & 31;
  const int half = lane >> 4, lh = lane & 15;
  const int bh = blockIdx.x;                    // 0..B*H-1
  const int bi = bh >> 4, hi = bh & (Hh - 1);
  const int q0 = (blockIdx.y * 8 + wave) * 16;
  const u16* qp = Q  + ((long)bh * Ss + q0) * DKk;
  const u16* kp = Km + (long)bh * Ss * DKk;
  const u16* vp = Vt + (long)bh * DKk * Ss;

  Frag aq0, aq1;   // Q tile as two A fragments (DK=64 -> k-steps 0 and 32)
  {
    const u16* ap = qp + lh * DKk;
    aq0.q[0] = ldg16(ap + half * 8);
    aq0.q[1] = ldg16(ap + 16 + half * 8);
    aq1.q[0] = ldg16(ap + 32 + half * 8);
    aq1.q[1] = ldg16(ap + 48 + half * 8);
  }

  float mrow[8], lrow[8];
  v8f cacc[4] = {};
  #pragma unroll
  for (int r = 0; r < 8; ++r) { mrow[r] = -3.0e38f; lrow[r] = 0.f; }
  const float scale = 0.125f;   // 1/sqrt(64)

  for (int s0 = 0; s0 < Ss; s0 += 32) {
    v8f sacc[2] = {};
    #pragma unroll
    for (int nt = 0; nt < 2; ++nt) {            // keys s0+nt*16 .. +15
      const u16* kr = kp + (long)(s0 + nt * 16 + lh) * DKk;   // B = K^T: contiguous per lane
      Frag kb0, kb1;
      kb0.q[0] = ldg16(kr + half * 8);
      kb0.q[1] = ldg16(kr + 16 + half * 8);
      kb1.q[0] = ldg16(kr + 32 + half * 8);
      kb1.q[1] = ldg16(kr + 48 + half * 8);
      sacc[nt] = wmma_bf16(aq0, kb0, sacc[nt]);
      sacc[nt] = wmma_bf16(aq1, kb1, sacc[nt]);
    }
    // online softmax: rows live across lanes 0-15 (half 0) / 16-31 (half 1)
    float corr[8];
    #pragma unroll
    for (int r = 0; r < 8; ++r) {
      float s0v = sacc[0][r] * scale;
      float s1v = sacc[1][r] * scale;
      float vmax = fmaxf(s0v, s1v);
      vmax = fmaxf(vmax, __shfl_xor(vmax, 1, 32));
      vmax = fmaxf(vmax, __shfl_xor(vmax, 2, 32));
      vmax = fmaxf(vmax, __shfl_xor(vmax, 4, 32));
      vmax = fmaxf(vmax, __shfl_xor(vmax, 8, 32));
      float mnew = fmaxf(mrow[r], vmax);
      float p0 = __expf(s0v - mnew);
      float p1 = __expf(s1v - mnew);
      float psum = p0 + p1;
      psum += __shfl_xor(psum, 1, 32);
      psum += __shfl_xor(psum, 2, 32);
      psum += __shfl_xor(psum, 4, 32);
      psum += __shfl_xor(psum, 8, 32);
      float c = __expf(mrow[r] - mnew);
      corr[r] = c;
      lrow[r] = lrow[r] * c + psum;
      mrow[r] = mnew;
      ldsP[wave][(r + 8 * half) * 32 + lh]      = f2bf(p0);   // C-layout coords
      ldsP[wave][(r + 8 * half) * 32 + 16 + lh] = f2bf(p1);
    }
    #pragma unroll
    for (int t = 0; t < 4; ++t)
      #pragma unroll
      for (int r = 0; r < 8; ++r)
        cacc[t][r] *= corr[r];
    // reload P as A fragment (16x32, K = keys)
    Frag ap;
    ap.q[0] = *reinterpret_cast<const uint4*>(&ldsP[wave][lh * 32 + half * 8]);
    ap.q[1] = *reinterpret_cast<const uint4*>(&ldsP[wave][lh * 32 + 16 + half * 8]);
    #pragma unroll
    for (int t = 0; t < 4; ++t) {               // ctx dk tiles of 16
      const u16* vr = vp + (long)(t * 16 + lh) * Ss + s0;     // Vt: contiguous per lane
      Frag bv;
      bv.q[0] = ldg16(vr + half * 8);
      bv.q[1] = ldg16(vr + 16 + half * 8);
      cacc[t] = wmma_bf16(ap, bv, cacc[t]);
    }
  }

  #pragma unroll
  for (int t = 0; t < 4; ++t)
    #pragma unroll
    for (int r = 0; r < 8; ++r) {
      float o = cacc[t][r] / lrow[r];
      Ctx[((long)bi * Ss + q0 + r + 8 * half) * (Hh * DKk) + hi * DKk + t * 16 + lh] = f2bf(o);
    }
}

// ---------------------------------------------------------------------------
// out = LayerNorm(X + Yres) * g + be ; writes fp32 (+ optional bf16 copy)
// one 256-thread block per row of D=1024
// ---------------------------------------------------------------------------
template<bool WB>
__global__ void __launch_bounds__(256) add_ln(
    const float* __restrict__ X, const float* __restrict__ Yr,
    const float* __restrict__ g, const float* __restrict__ be,
    float* __restrict__ Of, u16* __restrict__ Ob)
{
  __shared__ float sh[2][8];
  const int row = blockIdx.x, tid = threadIdx.x;
  const int wave = tid >> 5, lane = tid & 31;
  const float* xr = X + (long)row * Dd;
  const float* yr = Yr + (long)row * Dd;
  float v[4], s = 0.f, ss = 0.f;
  #pragma unroll
  for (int i = 0; i < 4; ++i) {
    float t = xr[tid + i * 256] + yr[tid + i * 256];
    v[i] = t; s += t; ss += t * t;
  }
  #pragma unroll
  for (int off = 1; off < 32; off <<= 1) {
    s  += __shfl_xor(s,  off, 32);
    ss += __shfl_xor(ss, off, 32);
  }
  if (lane == 0) { sh[0][wave] = s; sh[1][wave] = ss; }
  __syncthreads();
  float ts = 0.f, tss = 0.f;
  #pragma unroll
  for (int w = 0; w < 8; ++w) { ts += sh[0][w]; tss += sh[1][w]; }
  const float mu  = ts * (1.f / Dd);
  const float var = tss * (1.f / Dd) - mu * mu;
  const float rs  = rsqrtf(var + 1e-5f);
  #pragma unroll
  for (int i = 0; i < 4; ++i) {
    int c = tid + i * 256;
    float o = (v[i] - mu) * rs * g[c] + be[c];
    Of[(long)row * Dd + c] = o;
    if constexpr (WB) Ob[(long)row * Dd + c] = f2bf(o);
  }
}

// ---------------------------------------------------------------------------
extern "C" void kernel_launch(void* const* d_in, const int* in_sizes, int n_in,
                              void* d_out, int out_size, void* d_ws, size_t ws_size,
                              hipStream_t stream) {
  const float* src  = (const float*)d_in[0];
  const float* Wq   = (const float*)d_in[1];
  const float* bq   = (const float*)d_in[2];
  const float* Wk   = (const float*)d_in[3];
  const float* bk   = (const float*)d_in[4];
  const float* Wv   = (const float*)d_in[5];
  const float* bv   = (const float*)d_in[6];
  const float* Wo   = (const float*)d_in[7];
  const float* bo   = (const float*)d_in[8];
  const float* ln1g = (const float*)d_in[9];
  const float* ln1b = (const float*)d_in[10];
  const float* W1   = (const float*)d_in[11];
  const float* b1   = (const float*)d_in[12];
  const float* W2   = (const float*)d_in[13];
  const float* b2   = (const float*)d_in[14];
  const float* ln2g = (const float*)d_in[15];
  const float* ln2b = (const float*)d_in[16];

  size_t off = 0;
  auto walloc = [&](size_t bytes) -> void* {
    void* p = (char*)d_ws + off;
    off += (bytes + 255) & ~(size_t)255;
    return p;
  };
  u16*   srcb = (u16*)walloc((size_t)Mrows * Dd * 2);
  u16*   wqt  = (u16*)walloc((size_t)Hh * Dd * DKk * 2);  // [H*DK, D] = Wq^T per head
  u16*   wkt  = (u16*)walloc((size_t)Hh * Dd * DKk * 2);
  u16*   wvt  = (u16*)walloc((size_t)Hh * Dd * DKk * 2);
  u16*   wot  = (u16*)walloc((size_t)Dd * Hh * DKk * 2);  // [D, H*DK] = Wo^T
  u16*   w1t  = (u16*)walloc((size_t)Ff * Dd * 2);        // [F, D]    = W1^T
  u16*   w2t  = (u16*)walloc((size_t)Dd * Ff * 2);        // [D, F]    = W2^T
  u16*   qb   = (u16*)walloc((size_t)Mrows * Dd * 2);     // [B,H,S,DK]
  u16*   kbuf = (u16*)walloc((size_t)Mrows * Dd * 2);     // [B,H,S,DK]
  u16*   vtb  = (u16*)walloc((size_t)Mrows * Dd * 2);     // [B,H,DK,S]
  u16*   ctxb = (u16*)walloc((size_t)Mrows * Dd * 2);     // [B,S,H*DK]
  u16*   xb   = (u16*)walloc((size_t)Mrows * Dd * 2);     // ln1 out bf16
  u16*   hb   = (u16*)walloc((size_t)Mrows * Ff * 2);     // ffn hidden bf16
  float* yb   = (float*)walloc((size_t)Mrows * Dd * 4);   // gemm fp32 scratch
  float* xf   = (float*)walloc((size_t)Mrows * Dd * 4);   // ln1 out fp32

  dim3 blk(256);
  cast_f32_bf16<<<dim3(Mrows * Dd / 256), blk, 0, stream>>>(src, srcb, Mrows * Dd);

  dim3 tblk(32, 8);
  // qkv weights: batch=H, per-head [D,DK] -> [DK,D]; flat result = [H*DK, D]
  transpose_cast<<<dim3(DKk / 32, Dd / 32, Hh), tblk, 0, stream>>>(Wq, wqt, Dd, DKk);
  transpose_cast<<<dim3(DKk / 32, Dd / 32, Hh), tblk, 0, stream>>>(Wk, wkt, Dd, DKk);
  transpose_cast<<<dim3(DKk / 32, Dd / 32, Hh), tblk, 0, stream>>>(Wv, wvt, Dd, DKk);
  transpose_cast<<<dim3(Dd / 32, Hh * DKk / 32, 1), tblk, 0, stream>>>(Wo, wot, Hh * DKk, Dd);
  transpose_cast<<<dim3(Ff / 32, Dd / 32, 1), tblk, 0, stream>>>(W1, w1t, Dd, Ff);
  transpose_cast<<<dim3(Dd / 32, Ff / 32, 1), tblk, 0, stream>>>(W2, w2t, Ff, Dd);

  // GEMM grid: (N/64, M/256) with 8 waves x 32 rows per block
  // q/k/v projections: M=8192, N=1024, K=1024
  gemm_bf16_wmma<1><<<dim3(16, 32), blk, 0, stream>>>(srcb, Dd, wqt, bq, nullptr, qb,   Hh * DKk, Dd);
  gemm_bf16_wmma<1><<<dim3(16, 32), blk, 0, stream>>>(srcb, Dd, wkt, bk, nullptr, kbuf, Hh * DKk, Dd);
  gemm_bf16_wmma<4><<<dim3(16, 32), blk, 0, stream>>>(srcb, Dd, wvt, bv, nullptr, vtb,  Hh * DKk, Dd);
  // attention
  attn_flash<<<dim3(Bb * Hh, Ss / 128), blk, 0, stream>>>(qb, kbuf, vtb, ctxb);
  // output projection: ctx @ Wo + bo -> fp32
  gemm_bf16_wmma<3><<<dim3(16, 32), blk, 0, stream>>>(ctxb, Hh * DKk, wot, bo, yb, nullptr, Dd, Hh * DKk);
  // ln1: x = LN(src + attn_out)
  add_ln<true><<<dim3(Mrows), blk, 0, stream>>>(src, yb, ln1g, ln1b, xf, xb);
  // ffn: relu(x @ W1 + b1) @ W2 + b2
  gemm_bf16_wmma<2><<<dim3(64, 32), blk, 0, stream>>>(xb, Dd, w1t, b1, nullptr, hb, Ff, Dd);
  gemm_bf16_wmma<3><<<dim3(16, 32), blk, 0, stream>>>(hb, Ff, w2t, b2, yb, nullptr, Dd, Ff);
  // ln2: out = LN(x + ff)
  add_ln<false><<<dim3(Mrows), blk, 0, stream>>>(xf, yb, ln2g, ln2b, (float*)d_out, nullptr);
}
